// Net_33406255628726
// MI455X (gfx1250) — matrix operand order repaired and checked by reading, not compiled, not run on hardware
//
#include <hip/hip_runtime.h>
#include <hip/hip_bf16.h>

typedef __attribute__((ext_vector_type(16))) _Float16 v16h;
typedef __attribute__((ext_vector_type(4)))  _Float16 v4h;
typedef __attribute__((ext_vector_type(2)))  _Float16 v2h;
typedef __attribute__((ext_vector_type(8)))  float    v8f;

union V16 { v16h v; v4h q[4]; v2h d[8]; };

__device__ __forceinline__ float lrelu(float v) { return v > 0.f ? v : 0.01f * v; }

// ---------------------------------------------------------------------------
// Fused CNN per image (one 4-wave workgroup / image), WMMA f16 implicit GEMM.
// conv1: K re-tiled as 33 kernel-rows x 16 kx-slots (taps 11..15 = zero wt),
//        padded image in LDS -> fragment build = aligned b64 LDS loads only.
// conv2: 50 kernel-rows x 8 kx-slots (taps 5..7 zero), padded 11x16 pool img.
// Weights pre-swizzled into per-lane WMMA B-fragment layout in LDS.
// ---------------------------------------------------------------------------

// phase 1 (conv1): padded image + swizzled conv1 weights
struct Phase1 {
    alignas(16) _Float16 imgp[3 * 4556 + 16];   // [3][67][68] + overrun tail
    alignas(16) _Float16 w1f[17 * 32 * 16];     // [kc][lane][16] B fragments
    int kr1[34];                                // row -> c*4556 + ky*68
};
// phase 2 (conv2): padded pooled image + swizzled conv2 weights
struct Phase2 {
    alignas(16) _Float16 p1p[1792];             // [10][11][16] padded pool
    alignas(16) _Float16 w2f[13 * 32 * 16];     // [kc][lane][16] B fragments
    int kr2[52];                                // row -> c*176 + ky*16
};
union SMem { Phase1 p1; Phase2 p2; };

__global__ __launch_bounds__(128) void cnn_kernel(
    const float* __restrict__ im,   // [B,3,64,64]
    const float* __restrict__ w1,   // [10,3,11,11]
    const float* __restrict__ w2,   // [16,10,5,5]
    const float* __restrict__ b2,   // [16]
    const float* __restrict__ lw,   // [2,64]
    const float* __restrict__ lb,   // [2]
    float* __restrict__ imgfeat)    // d_out + B*20, [B,2]
{
    __shared__ SMem  sm;
    __shared__ float s_c1[240 * 16];            // conv1 out [Mpad][Npad] f32
    __shared__ float s_feat[64];

    const int tid  = threadIdx.x;
    const int lane = tid & 31;
    const int wv   = tid >> 5;
    const int b    = blockIdx.x;
    const int mrow = lane & 15;                 // M row (A) / N col (B/C)
    const int khi  = lane >> 4;                 // K-half selector

    // ---- phase-1 fill: padded image, conv1 B fragments, row table --------
    const float* imb = im + (size_t)b * (3 * 64 * 64);
    for (int i = tid; i < 3 * 4556 + 16; i += 128) {
        float v = 0.f;
        if (i < 3 * 4556) {
            int c = i / 4556, rem = i - c * 4556;
            int iy = rem / 68 - 2, ix = rem % 68 - 2;
            if ((unsigned)iy < 64u && (unsigned)ix < 64u)
                v = imb[c * 4096 + iy * 64 + ix];
        }
        sm.p1.imgp[i] = (_Float16)v;
    }
    for (int i = tid; i < 17 * 512; i += 128) {
        int kc = i >> 9, L = (i >> 4) & 31, h = i & 15;
        int K  = kc * 32 + ((h >> 3) << 4) + ((L >> 4) << 3) + (h & 7);     // B-layout K index
        int r  = K >> 4, kx = K & 15, col = L & 15;
        float v = 0.f;
        if (r < 33 && kx < 11 && col < 10)
            v = w1[col * 363 + (r / 11) * 121 + (r % 11) * 11 + kx];
        sm.p1.w1f[i] = (_Float16)v;
    }
    for (int r = tid; r < 34; r += 128)
        sm.p1.kr1[r] = (r < 33) ? ((r / 11) * 4556 + (r % 11) * 68) : 0;
    __syncthreads();

    // ---- conv1: M=225 out px (15 tiles), N=10->16 ch, K=33x16=528 --------
    for (int mt = wv; mt < 15; mt += 4) {
        int M  = mt * 16 + mrow; if (M > 224) M = 224;
        int oy = M / 15, ox = M % 15;
        int pixbase = oy * 4 * 68 + ox * 4 + 8 * khi;   // multiple of 4 halves
        v8f acc = {};
        for (int kc = 0; kc < 17; ++kc) {
            int off0 = sm.p1.kr1[2 * kc]     + pixbase;
            int off1 = sm.p1.kr1[2 * kc + 1] + pixbase;
            const v4h* pa0 = (const v4h*)&sm.p1.imgp[off0];
            const v4h* pa1 = (const v4h*)&sm.p1.imgp[off1];
            const v4h* pb  = (const v4h*)&sm.p1.w1f[(kc * 32 + lane) * 16];
            V16 a, bb;
            a.q[0] = pa0[0]; a.q[1] = pa0[1];           // row r0, kx 8*khi..+7
            a.q[2] = pa1[0]; a.q[3] = pa1[1];           // row r1
            bb.q[0] = pb[0]; bb.q[1] = pb[1]; bb.q[2] = pb[2]; bb.q[3] = pb[3];
            acc = __builtin_amdgcn_wmma_f32_16x16x32_f16(false, a.v, false, bb.v,
                                                         (short)0, acc, false, false);
        }
#pragma unroll
        for (int v = 0; v < 8; ++v) {                   // D: M = v + 8*khi
            int Mg = mt * 16 + v + (khi << 3);
            s_c1[Mg * 16 + mrow] = lrelu(acc[v]);
        }
    }
    __syncthreads();

    // ---- phase-2 fill: pool 3x3 s2 into padded [10][11][16], conv2 wts ---
    for (int i = tid; i < 1792; i += 128) {
        float v = 0.f;
        if (i < 1760) {
            int c = i / 176, rem = i % 176;
            int py = rem / 16 - 2, px = rem % 16 - 2;
            if ((unsigned)py < 7u && (unsigned)px < 7u) {
                float m = -1e30f;
#pragma unroll
                for (int dy = 0; dy < 3; ++dy)
#pragma unroll
                    for (int dx = 0; dx < 3; ++dx)
                        m = fmaxf(m, s_c1[((py * 2 + dy) * 15 + (px * 2 + dx)) * 16 + c]);
                v = m;
            }
        }
        sm.p2.p1p[i] = (_Float16)v;
    }
    for (int i = tid; i < 13 * 512; i += 128) {
        int kc = i >> 9, L = (i >> 4) & 31, h = i & 15;
        int K  = kc * 32 + ((h >> 3) << 4) + ((L >> 4) << 3) + (h & 7);
        int r  = K >> 3, kx = K & 7, col = L & 15;
        float v = 0.f;
        if (r < 50 && kx < 5)
            v = w2[col * 250 + (r / 5) * 25 + (r % 5) * 5 + kx];
        sm.p2.w2f[i] = (_Float16)v;
    }
    for (int r = tid; r < 52; r += 128)
        sm.p2.kr2[r] = (r < 50) ? ((r / 5) * 176 + (r % 5) * 16) : 0;
    __syncthreads();

    // ---- conv2: one 16x16 tile, K=50x8=400 (13 chunks); wave 0 only ------
    if (wv == 0) {
        int oy = mrow >> 2, ox = mrow & 3;              // 4x4 out pixels
        int pixbase = oy * 2 * 16 + ox * 2;             // multiple of 2 halves
        v8f acc = {};
        for (int kc = 0; kc < 13; ++kc) {
            int rA = 4 * kc + khi;                      // rows per A layout
            int offA = sm.p2.kr2[rA]     + pixbase;
            int offB = sm.p2.kr2[rA + 2] + pixbase;
            const v2h* pa0 = (const v2h*)&sm.p2.p1p[offA];
            const v2h* pa1 = (const v2h*)&sm.p2.p1p[offB];
            const v4h* pb  = (const v4h*)&sm.p2.w2f[(kc * 32 + lane) * 16];
            V16 a, bb;
            a.d[0] = pa0[0]; a.d[1] = pa0[1]; a.d[2] = pa0[2]; a.d[3] = pa0[3];
            a.d[4] = pa1[0]; a.d[5] = pa1[1]; a.d[6] = pa1[2]; a.d[7] = pa1[3];
            bb.q[0] = pb[0]; bb.q[1] = pb[1]; bb.q[2] = pb[2]; bb.q[3] = pb[3];
            acc = __builtin_amdgcn_wmma_f32_16x16x32_f16(false, a.v, false, bb.v,
                                                         (short)0, acc, false, false);
        }
        float bias = b2[mrow];
        float v0 = lrelu(acc[0] + bias), v1 = lrelu(acc[1] + bias);
        float v2 = lrelu(acc[2] + bias), v3 = lrelu(acc[3] + bias);
        float v4 = lrelu(acc[4] + bias), v5 = lrelu(acc[5] + bias);
        float v6 = lrelu(acc[6] + bias), v7 = lrelu(acc[7] + bias);
        // maxpool 2x2: lane holds pixel rows {0,1} (khi=0) or {2,3} (khi=1)
        float pA = fmaxf(fmaxf(v0, v1), fmaxf(v4, v5)); // window (khi, 0)
        float pB = fmaxf(fmaxf(v2, v3), fmaxf(v6, v7)); // window (khi, 1)
        s_feat[mrow * 4 + khi * 2 + 0] = pA;
        s_feat[mrow * 4 + khi * 2 + 1] = pB;
    }
    __syncthreads();

    // ---- linear 64->2 + sigmoid ------------------------------------------
    if (tid < 2) {
        float acc = lb[tid];
        for (int j = 0; j < 64; ++j) acc += s_feat[j] * lw[tid * 64 + j];
        imgfeat[b * 2 + tid] = 1.f / (1.f + __expf(-acc));
    }
}

// ---------------------------------------------------------------------------
// GCN: per-graph workgroup (20 nodes, 380 edges). Deterministic (no atomics):
// each of 20 lanes scans the edge list in fixed order with masked FMAs.
// ---------------------------------------------------------------------------
__global__ __launch_bounds__(32) void gcn_kernel(
    const float* __restrict__ x,        // [N,2]
    const int*   __restrict__ ei,       // [2,E]
    int Etot,
    const float* __restrict__ imgfeat,  // [B,2]
    const float* __restrict__ g1w, const float* __restrict__ g1b,
    const float* __restrict__ g2w, const float* __restrict__ g2b,
    float* __restrict__ out)            // [B,20]
{
    constexpr int NN = 20, EPG = NN * (NN - 1);  // 380
    __shared__ int   s_e[EPG];
    __shared__ float s_xin[NN * 4];
    __shared__ float s_x1[NN * 8];
    __shared__ float s_dis1[NN], s_dis2[NN];
    __shared__ int   s_deg[NN];
    __shared__ float s_x2[NN];
    __shared__ float s_h2[NN];
    __shared__ float s_g1w[32], s_g1b[8], s_g2w[8], s_g2b[1];

    const int tid = threadIdx.x;
    const int b   = blockIdx.x;
    const int e0  = b * EPG;
    const int n0  = b * NN;

    for (int e = tid; e < EPG; e += 32) {
        int s = ei[e0 + e] - n0;
        int d = ei[Etot + e0 + e] - n0;
        s_e[e] = (s & 255) | (d << 8);
    }
    s_g1w[tid] = g1w[tid];
    if (tid < 8) { s_g1b[tid] = g1b[tid]; s_g2w[tid] = g2w[tid]; }
    if (tid == 0) s_g2b[0] = g2b[0];
    for (int t = tid; t < NN * 4; t += 32) {
        int i = t >> 2, k = t & 3;
        s_xin[t] = (k < 2) ? x[(n0 + i) * 2 + k] : imgfeat[b * 2 + (k - 2)];
    }
    __syncthreads();

    for (int t = tid; t < NN * 8; t += 32) {
        int i = t >> 3, c = t & 7;
        float a = 0.f;
#pragma unroll
        for (int k = 0; k < 4; ++k) a += s_xin[i * 4 + k] * s_g1w[c * 4 + k];
        s_x1[t] = a;
    }
    if (tid < NN) {
        int cnt = 0;
        for (int e = 0; e < EPG; ++e) cnt += ((s_e[e] >> 8) == tid);
        s_deg[tid]  = cnt;
        s_dis1[tid] = rsqrtf((float)cnt + 1.f);  // layer1: fill = 1
        s_dis2[tid] = rsqrtf((float)cnt + 2.f);  // layer2: fill = 2 (improved)
    }
    __syncthreads();

    if (tid < NN) {
        const int i = tid;
        float di = s_dis1[i];
        float s[8] = {};
        for (int e = 0; e < EPG; ++e) {
            int p = s_e[e];
            int ls = p & 255, ld = p >> 8;
            float w = (ld == i) ? di * s_dis1[ls] : 0.f;
#pragma unroll
            for (int c = 0; c < 8; ++c) s[c] += w * s_x1[ls * 8 + c];
        }
        float selfn = di * di;                       // self-loop norm, ew = 1
        float cnt   = (float)s_deg[i] + 1.f;         // mean divisor
        float x2 = 0.f;
#pragma unroll
        for (int c = 0; c < 8; ++c) {
            float h = (s[c] + selfn * s_x1[i * 8 + c]) / cnt + s_g1b[c];
            h = lrelu(h);
            x2 += h * s_g2w[c];                      // x2 = h1 @ g2_w.T
        }
        s_x2[i] = x2;
    }
    __syncthreads();

    if (tid < NN) {
        const int i = tid;
        float di = s_dis2[i];
        float m = -3.0e38f;
        for (int e = 0; e < EPG; ++e) {
            int p = s_e[e];
            int ls = p & 255, ld = p >> 8;
            float msg = (ld == i) ? s_x2[ls] * di * s_dis2[ls] : -3.0e38f;
            m = fmaxf(m, msg);
        }
        m = fmaxf(m, 2.f * s_x2[i] * di * di);       // self loop, ew = 2
        if (!(m > -3.0e38f && m < 3.0e38f)) m = 0.f; // isfinite guard
        s_h2[i] = m + s_g2b[0];
    }
    __syncthreads();

    if (tid < NN) {
        float mx = s_h2[0];
        for (int j = 1; j < NN; ++j) mx = fmaxf(mx, s_h2[j]);
        float sum = 0.f;
        for (int j = 0; j < NN; ++j) sum += __expf(s_h2[j] - mx);
        out[n0 + tid] = __expf(s_h2[tid] - mx) / sum;
    }
}

extern "C" void kernel_launch(void* const* d_in, const int* in_sizes, int n_in,
                              void* d_out, int out_size, void* d_ws, size_t ws_size,
                              hipStream_t stream) {
    const float* im  = (const float*)d_in[0];
    const float* x   = (const float*)d_in[1];
    const int*   ei  = (const int*)d_in[2];
    const float* w1  = (const float*)d_in[3];
    const float* w2  = (const float*)d_in[4];
    const float* b2  = (const float*)d_in[5];
    const float* lw  = (const float*)d_in[6];
    const float* lb  = (const float*)d_in[7];
    const float* g1w = (const float*)d_in[8];
    const float* g1b = (const float*)d_in[9];
    const float* g2w = (const float*)d_in[10];
    const float* g2b = (const float*)d_in[11];
    float* out = (float*)d_out;

    const int B    = in_sizes[0] / (3 * 64 * 64);   // 2048
    const int Etot = in_sizes[2] / 2;               // B * 380
    float* imgfeat = out + (size_t)B * 20;          // second tuple output

    cnn_kernel<<<B, 128, 0, stream>>>(im, w1, w2, b2, lw, lb, imgfeat);
    gcn_kernel<<<B, 32, 0, stream>>>(x, ei, Etot, imgfeat, g1w, g1b, g2w, g2b, out);
}
